// AttnBlock_1211180777548
// MI455X (gfx1250) — compile-verified, matrix operand order checked by reading, not run on hardware
//
// CDNA5 / gfx1250 attention block (GroupNorm -> QKV -> softmax(QK^T*scale) -> AV -> proj + residual)
// Compute-bound (~86 GFLOP vs ~36MB mandatory HBM traffic at 23.3 TB/s): all GEMMs run on
// v_wmma_f32_16x16x32_bf16 (wave32 WMMA), with double-buffered GLOBAL_LOAD_ASYNC_TO_LDS_B128
// DMA staging (ASYNCcnt) so tile k+1's HBM->LDS transfer overlaps tile k's matrix math.
// Workspace requirement ~234 MB.
#include <hip/hip_runtime.h>
#include <hip/hip_bf16.h>
#include <math.h>

typedef __bf16 bf16_t;
typedef __attribute__((ext_vector_type(16))) __bf16 v16bf;
typedef __attribute__((ext_vector_type(8)))  __bf16 v8bf;
typedef __attribute__((ext_vector_type(8)))  float  v8f;

struct __align__(16) Chunk16 { unsigned long long lo, hi; };

constexpr int BATCH = 2;
constexpr int CH    = 512;
constexpr int NSP   = 4096;          // 64*64 tokens
constexpr int NGRP  = 32;
constexpr int CPG   = CH / NGRP;     // 16 channels per group
constexpr float GN_EPS = 1e-6f;

__device__ __forceinline__ bf16_t f2bf(float f) {
  unsigned u = __builtin_bit_cast(unsigned, f);
  unsigned r = (u + 0x7FFFu + ((u >> 16) & 1u)) >> 16;
  unsigned short h = (unsigned short)r;
  return __builtin_bit_cast(bf16_t, h);
}

// ---- gfx1250 async DMA: global -> LDS, 16B per lane, tracked by ASYNCcnt ----
__device__ __forceinline__ void async_copy16(const void* g, void* l) {
  // low 32 bits of a generic LDS pointer are the LDS byte address (ISA 10.2 aperture rule)
  unsigned lds = (unsigned)(size_t)l;
  unsigned long long ga = (unsigned long long)(size_t)g;
  asm volatile("global_load_async_to_lds_b128 %0, %1, off"
               :: "v"(lds), "v"(ga) : "memory");
}
__device__ __forceinline__ void wait_async0() {
  asm volatile("s_wait_asynccnt 0x0" ::: "memory");
}

// ---------------- GroupNorm ----------------
__global__ __launch_bounds__(256)
void gn_stats_kernel(const float* __restrict__ x, float* __restrict__ stats) {
  const int g = blockIdx.x, b = blockIdx.y;
  const float* xp = x + ((size_t)b * CH + (size_t)g * CPG) * NSP; // contiguous group slab
  const int NE = CPG * NSP; // 65536
  float s = 0.f, s2 = 0.f;
  for (int i = threadIdx.x; i < NE; i += 256) { float v = xp[i]; s += v; s2 += v * v; }
  __shared__ float rs[256], rq[256];
  rs[threadIdx.x] = s; rq[threadIdx.x] = s2;
  __syncthreads();
  for (int off = 128; off > 0; off >>= 1) {
    if (threadIdx.x < off) { rs[threadIdx.x] += rs[threadIdx.x + off]; rq[threadIdx.x] += rq[threadIdx.x + off]; }
    __syncthreads();
  }
  if (threadIdx.x == 0) {
    float mu  = rs[0] / (float)NE;
    float var = rq[0] / (float)NE - mu * mu;
    stats[(b * NGRP + g) * 2 + 0] = mu;
    stats[(b * NGRP + g) * 2 + 1] = rsqrtf(var + GN_EPS);
  }
}

__global__ __launch_bounds__(256)
void gn_apply_kernel(const float* __restrict__ x, const float* __restrict__ gamma,
                     const float* __restrict__ beta, const float* __restrict__ stats,
                     bf16_t* __restrict__ hn) {
  const int c = blockIdx.x, b = blockIdx.y;
  const int g = c / CPG;
  const float mu = stats[(b * NGRP + g) * 2 + 0];
  const float rstd = stats[(b * NGRP + g) * 2 + 1];
  const float ga = gamma[c] * rstd;
  const float be = beta[c] - mu * ga;            // (x-mu)*rstd*gamma + beta = x*ga + be
  const float* xp = x + ((size_t)b * CH + c) * NSP;
  bf16_t* hp = hn + (size_t)b * NSP * CH + c;    // hn layout: [b][pos][c]
  for (int p = threadIdx.x; p < NSP; p += 256)
    hp[(size_t)p * CH] = f2bf(xp[p] * ga + be);
}

// ---------------- f32 -> bf16 weight conversion ----------------
__global__ __launch_bounds__(256)
void cvt_kernel(const float* __restrict__ src, bf16_t* __restrict__ dst, int n) {
  int i = blockIdx.x * 256 + threadIdx.x;
  if (i < n) dst[i] = f2bf(src[i]);
}

// ---------------- Tiled WMMA GEMM: D = A[MxK] * B[NxK]^T (+bias) ----------------
// Block tile 128x128, 8 waves in a 2(M) x 4(N) grid; each wave: 4x2 subtiles of 16x16.
// K-step 32; double-buffered async HBM->LDS staging.
enum { EPI_BF16 = 0, EPI_TRANS = 1, EPI_F32 = 2, EPI_RESID = 3 };

template<int EPI>
__global__ __launch_bounds__(256)
void gemm_bt_kernel(const bf16_t* __restrict__ A, const bf16_t* __restrict__ Bm,
                    const float* __restrict__ bias, void* __restrict__ outp,
                    const float* __restrict__ resid, float alpha,
                    int Kd, int ldOut,
                    size_t strideA, size_t strideB, size_t strideOut, size_t strideResid) {
  constexpr int BM = 128, BN = 128, BK = 32;
  __shared__ Chunk16 sA[2][BM * 5];  // 32 bf16 data + 8 bf16 pad per row (16B-aligned rows)
  __shared__ Chunk16 sB[2][BN * 5];

  const int tid = threadIdx.x;
  const int wave = tid >> 5, lane = tid & 31;
  const int waveM = wave >> 2;         // 0..1  -> 64-row strip
  const int waveN = wave & 3;          // 0..3  -> 32-col strip
  const int lane16 = lane & 15, laneHi = lane >> 4;

  const size_t z = blockIdx.z;
  const bf16_t* Ab = A + z * strideA;
  const bf16_t* Bb = Bm + z * strideB;
  bf16_t* outB = (bf16_t*)outp + z * strideOut;
  float*  outF = (float*)outp + z * strideOut;
  const float* rz = (EPI == EPI_RESID) ? (resid + z * strideResid) : nullptr;

  const int rowA0 = blockIdx.y * BM;
  const int colB0 = blockIdx.x * BN;

  // staging coordinates for this thread (2 x 16B per matrix per K-step)
  const int r0 = (tid * 2) >> 2,       kc0 = (tid * 2) & 3;
  const int r1 = (tid * 2 + 1) >> 2,   kc1 = (tid * 2 + 1) & 3;

  const v8f vzero = {0.f, 0.f, 0.f, 0.f, 0.f, 0.f, 0.f, 0.f};
  v8f acc[4][2];
#pragma unroll
  for (int i = 0; i < 4; ++i)
#pragma unroll
    for (int j = 0; j < 2; ++j) acc[i][j] = vzero;

  const int nIter = Kd / BK;

  // prologue: kick off DMA for the first K tile
  {
    const int kk = 0;
    async_copy16(Ab + (size_t)(rowA0 + r0) * Kd + kk + kc0 * 8, &sA[0][r0 * 5 + kc0]);
    async_copy16(Ab + (size_t)(rowA0 + r1) * Kd + kk + kc1 * 8, &sA[0][r1 * 5 + kc1]);
    async_copy16(Bb + (size_t)(colB0 + r0) * Kd + kk + kc0 * 8, &sB[0][r0 * 5 + kc0]);
    async_copy16(Bb + (size_t)(colB0 + r1) * Kd + kk + kc1 * 8, &sB[0][r1 * 5 + kc1]);
  }

  for (int it = 0; it < nIter; ++it) {
    wait_async0();        // this wave's DMA for buffer it&1 has landed in LDS
    __syncthreads();      // all waves' DMA landed; all waves done reading buffer (it+1)&1

    if (it + 1 < nIter) { // overlap: DMA for tile it+1 runs under this tile's WMMAs
      const int kk = (it + 1) * BK, nb = (it + 1) & 1;
      async_copy16(Ab + (size_t)(rowA0 + r0) * Kd + kk + kc0 * 8, &sA[nb][r0 * 5 + kc0]);
      async_copy16(Ab + (size_t)(rowA0 + r1) * Kd + kk + kc1 * 8, &sA[nb][r1 * 5 + kc1]);
      async_copy16(Bb + (size_t)(colB0 + r0) * Kd + kk + kc0 * 8, &sB[nb][r0 * 5 + kc0]);
      async_copy16(Bb + (size_t)(colB0 + r1) * Kd + kk + kc1 * 8, &sB[nb][r1 * 5 + kc1]);
      if (it + 2 < nIter) // warm L2 for the tile after next (global_prefetch_b8)
        __builtin_prefetch(Ab + (size_t)(rowA0 + (tid >> 1)) * Kd + kk + BK, 0, 3);
    }

    const bf16_t* sAh = (const bf16_t*)sA[it & 1];
    const bf16_t* sBh = (const bf16_t*)sB[it & 1];

    // ---- B fragments: lanes hold N columns; K split per ISA 16-bit layout ----
    v16bf bfrag[2];
#pragma unroll
    for (int nsub = 0; nsub < 2; ++nsub) {
      const int rB = waveN * 32 + nsub * 16 + lane16;
      v8bf lo = *(const v8bf*)(sBh + (size_t)rB * 40 + laneHi * 8);       // K 0-7 / 8-15
      v8bf hi = *(const v8bf*)(sBh + (size_t)rB * 40 + 16 + laneHi * 8);  // K 16-23 / 24-31
      bfrag[nsub] = __builtin_shufflevector(lo, hi, 0,1,2,3,4,5,6,7,8,9,10,11,12,13,14,15);
    }
#pragma unroll
    for (int msub = 0; msub < 4; ++msub) {
      const int rA = waveM * 64 + msub * 16 + lane16;
      v8bf lo = *(const v8bf*)(sAh + (size_t)rA * 40 + laneHi * 8);
      v8bf hi = *(const v8bf*)(sAh + (size_t)rA * 40 + 16 + laneHi * 8);
      v16bf afrag = __builtin_shufflevector(lo, hi, 0,1,2,3,4,5,6,7,8,9,10,11,12,13,14,15);
#pragma unroll
      for (int nsub = 0; nsub < 2; ++nsub) {
        acc[msub][nsub] = __builtin_amdgcn_wmma_f32_16x16x32_bf16(
            false, afrag, false, bfrag[nsub], (short)0, acc[msub][nsub], false, false);
      }
    }
    __syncthreads();      // done reading buffer it&1 before iter it+1 issues into (it+2)&1
  }

  // ---- epilogue: C/D layout lane L -> (m = r + 8*(L>=16), n = L%16) ----
#pragma unroll
  for (int msub = 0; msub < 4; ++msub) {
#pragma unroll
    for (int nsub = 0; nsub < 2; ++nsub) {
      const int gcol = colB0 + waveN * 32 + nsub * 16 + lane16;
      const float bv = (EPI == EPI_F32) ? 0.f : (bias ? bias[gcol] : 0.f);
#pragma unroll
      for (int r = 0; r < 8; ++r) {
        const int grow = rowA0 + waveM * 64 + msub * 16 + r + 8 * laneHi;
        float val = (acc[msub][nsub][r] + bv) * alpha;
        if (EPI == EPI_BF16) {
          outB[(size_t)grow * ldOut + gcol] = f2bf(val);
        } else if (EPI == EPI_TRANS) {
          outB[(size_t)gcol * ldOut + grow] = f2bf(val);
        } else if (EPI == EPI_F32) {
          outF[(size_t)grow * ldOut + gcol] = val;
        } else { // EPI_RESID: d_out[b][o][p] = x[b][o][p] + proj
          const size_t idx = (size_t)gcol * ldOut + grow;
          outF[idx] = rz[idx] + val;
        }
      }
    }
  }
}

// ---------------- row softmax: S (f32) -> P (bf16) ----------------
__global__ __launch_bounds__(256)
void softmax_kernel(const float* __restrict__ S, bf16_t* __restrict__ P) {
  const int i = blockIdx.x, b = blockIdx.y;
  const float* sp = S + ((size_t)b * NSP + i) * NSP;
  bf16_t* pp = P + ((size_t)b * NSP + i) * NSP;
  __shared__ float red[256];
  float m = -3.4e38f;
  for (int j = threadIdx.x; j < NSP; j += 256) m = fmaxf(m, sp[j]);
  red[threadIdx.x] = m; __syncthreads();
  for (int off = 128; off > 0; off >>= 1) {
    if (threadIdx.x < off) red[threadIdx.x] = fmaxf(red[threadIdx.x], red[threadIdx.x + off]);
    __syncthreads();
  }
  m = red[0]; __syncthreads();
  float s = 0.f;
  for (int j = threadIdx.x; j < NSP; j += 256) s += __expf(sp[j] - m);
  red[threadIdx.x] = s; __syncthreads();
  for (int off = 128; off > 0; off >>= 1) {
    if (threadIdx.x < off) red[threadIdx.x] += red[threadIdx.x + off];
    __syncthreads();
  }
  const float inv = 1.f / red[0];
  for (int j = threadIdx.x; j < NSP; j += 256) pp[j] = f2bf(__expf(sp[j] - m) * inv);
}

// ---------------- host launch ----------------
extern "C" void kernel_launch(void* const* d_in, const int* in_sizes, int n_in,
                              void* d_out, int out_size, void* d_ws, size_t ws_size,
                              hipStream_t stream) {
  (void)in_sizes; (void)n_in; (void)out_size; (void)ws_size;
  const float* x  = (const float*)d_in[0];
  const float* gg = (const float*)d_in[1];
  const float* gb = (const float*)d_in[2];
  const float* wq = (const float*)d_in[3];
  const float* bq = (const float*)d_in[4];
  const float* wk = (const float*)d_in[5];
  const float* bk = (const float*)d_in[6];
  const float* wv = (const float*)d_in[7];
  const float* bv = (const float*)d_in[8];
  const float* wo = (const float*)d_in[9];
  const float* bo = (const float*)d_in[10];

  // workspace carve-up (~234 MB): stats + 5x 8MB bf16 acts + 4x 0.5MB bf16 weights
  // + 128MB f32 scores + 64MB bf16 probs
  char* wsp = (char*)d_ws;
  auto take = [&](size_t bytes) -> void* {
    void* p = (void*)wsp; wsp += (bytes + 255) & ~(size_t)255; return p;
  };
  const size_t ACT = (size_t)BATCH * NSP * CH;      // per-tensor activation elems
  const size_t SCO = (size_t)BATCH * NSP * NSP;     // score elems
  float*  stats = (float*)take((size_t)BATCH * NGRP * 2 * sizeof(float));
  bf16_t* hn  = (bf16_t*)take(ACT * 2);
  bf16_t* qb  = (bf16_t*)take(ACT * 2);
  bf16_t* kb  = (bf16_t*)take(ACT * 2);
  bf16_t* vT  = (bf16_t*)take(ACT * 2);             // V^T: [b][c][j]
  bf16_t* Ob  = (bf16_t*)take(ACT * 2);             // attn output [b][i][c]
  bf16_t* wqb = (bf16_t*)take((size_t)CH * CH * 2);
  bf16_t* wkb = (bf16_t*)take((size_t)CH * CH * 2);
  bf16_t* wvb = (bf16_t*)take((size_t)CH * CH * 2);
  bf16_t* wob = (bf16_t*)take((size_t)CH * CH * 2);
  float*  Sb  = (float*)take(SCO * sizeof(float));
  bf16_t* Pb  = (bf16_t*)take(SCO * 2);

  const float scale = 1.0f / sqrtf((float)CH);
  const size_t nCH = (size_t)NSP * CH;
  const size_t nnS = (size_t)NSP * NSP;
  const size_t cnT = (size_t)CH * NSP;

  // 1) GroupNorm
  gn_stats_kernel<<<dim3(NGRP, BATCH), 256, 0, stream>>>(x, stats);
  gn_apply_kernel<<<dim3(CH, BATCH), 256, 0, stream>>>(x, gg, gb, stats, hn);

  // 2) weights -> bf16
  const int WN = CH * CH;
  cvt_kernel<<<(WN + 255) / 256, 256, 0, stream>>>(wq, wqb, WN);
  cvt_kernel<<<(WN + 255) / 256, 256, 0, stream>>>(wk, wkb, WN);
  cvt_kernel<<<(WN + 255) / 256, 256, 0, stream>>>(wv, wvb, WN);
  cvt_kernel<<<(WN + 255) / 256, 256, 0, stream>>>(wo, wob, WN);

  // 3) QKV projections: [NSP x CH] = hn x W^T  (Q pre-scaled by C^-1/2)
  dim3 gQKV(CH / 128, NSP / 128, BATCH);
  gemm_bt_kernel<EPI_BF16 ><<<gQKV, 256, 0, stream>>>(hn, wqb, bq, qb, nullptr, scale,
      CH, CH, nCH, 0, nCH, 0);
  gemm_bt_kernel<EPI_BF16 ><<<gQKV, 256, 0, stream>>>(hn, wkb, bk, kb, nullptr, 1.0f,
      CH, CH, nCH, 0, nCH, 0);
  gemm_bt_kernel<EPI_TRANS><<<gQKV, 256, 0, stream>>>(hn, wvb, bv, vT, nullptr, 1.0f,
      CH, NSP, nCH, 0, cnT, 0);

  // 4) scores: S[b,i,j] = q_i . k_j   (M=N=4096, K=512)
  dim3 gS(NSP / 128, NSP / 128, BATCH);
  gemm_bt_kernel<EPI_F32><<<gS, 256, 0, stream>>>(qb, kb, nullptr, Sb, nullptr, 1.0f,
      CH, NSP, nCH, nCH, nnS, 0);

  // 5) softmax over j
  softmax_kernel<<<dim3(NSP, BATCH), 256, 0, stream>>>(Sb, Pb);

  // 6) O[b,i,c] = P x V  (A=P [NSPxNSP], B=V^T [CHxNSP], K=4096)
  dim3 gAV(CH / 128, NSP / 128, BATCH);
  gemm_bt_kernel<EPI_BF16><<<gAV, 256, 0, stream>>>(Pb, vT, nullptr, Ob, nullptr, 1.0f,
      NSP, CH, nnS, cnT, nCH, 0);

  // 7) out proj + residual, transposed store back to [b][c][p] f32
  dim3 gO(CH / 128, NSP / 128, BATCH);
  gemm_bt_kernel<EPI_RESID><<<gO, 256, 0, stream>>>(Ob, wob, bo, d_out, x, 1.0f,
      CH, NSP, nCH, 0, cnT, cnT);
}
